// GCNN_13632226197556
// MI455X (gfx1250) — matrix operand order repaired and checked by reading, not compiled, not run on hardware
//
#include <hip/hip_runtime.h>

#define N_PROT   50000
#define N_SUB    2000
#define D_PROT   256
#define D_SUB    256
#define D_PROJ   128
#define KERNEL_K 16
#define N_INT    64
#define NUM_WIN  (N_PROT - KERNEL_K + 1)   // 49985

// ---- workspace layout (float indices) ----
#define WS_G      0
#define WS_P      (WS_G + N_PROT)
#define WS_V1     (WS_P + N_PROT)          // Wp @ w1           (256)
#define WS_VP     (WS_V1 + D_PROT)         // Wp @ ones         (256)
#define WS_REACT  (WS_VP + D_PROT)         // reaction          (128)
#define WS_CONST  (WS_REACT + D_PROJ)      // [0]=gconst [1]=Pconst
#define WS_BVAL   (WS_CONST + 4)           // per-block argmax vals (256)
#define WS_SDELTA (WS_BVAL + 256)          // sub delta         (256)
#define WS_PDELTA (WS_SDELTA + D_SUB)      // prot delta        (256)
#define WS_INT    (WS_PDELTA + D_PROT)     // int area: BIDX[256], TOP at [256]

// ---- d_out layout ----
#define OUT_SUB   ((size_t)N_PROT * D_PROT)                 // 12,800,000
#define OUT_IDX   (OUT_SUB + (size_t)N_SUB * D_SUB)        // 13,312,000

typedef __attribute__((ext_vector_type(16))) _Float16 v16h;
typedef __attribute__((ext_vector_type(8)))  float    v8f;
typedef __attribute__((ext_vector_type(4)))  float    v4f;

// ------------------------------------------------------------------
// k0: precompute v1 = Wp@w1, vP = Wp@1, reaction, gconst, Pconst
// ------------------------------------------------------------------
__global__ __launch_bounds__(256) void k0_precompute(
    const float* __restrict__ sub, const int* __restrict__ idx,
    const float* __restrict__ Wp, const float* __restrict__ bp,
    const float* __restrict__ Ws, const float* __restrict__ bs,
    const float* __restrict__ wa, const float* __restrict__ ba,
    float* __restrict__ W)
{
    __shared__ float ssum[D_SUB];
    __shared__ float react[D_PROJ];
    __shared__ float red[256];
    const int t = threadIdx.x;

    // sum of the 64 gathered sub rows
    float s = 0.f;
    for (int j = 0; j < N_INT; ++j) s += sub[(size_t)idx[j] * D_SUB + t];
    ssum[t] = s;

    // v1[t] = Wp[t,:]·w1 ; vP[t] = sum(Wp[t,:])
    float a = 0.f, b = 0.f;
    const float* wrow = Wp + (size_t)t * D_PROJ;
    for (int n = 0; n < D_PROJ; ++n) { float w = wrow[n]; a += w * wa[n]; b += w; }
    W[WS_V1 + t] = a;
    W[WS_VP + t] = b;
    __syncthreads();

    // reaction = ssum @ Ws + 64*bs
    if (t < D_PROJ) {
        float r = 64.f * bs[t];
        for (int k = 0; k < D_SUB; ++k) r += ssum[k] * Ws[(size_t)k * D_PROJ + t];
        react[t] = r;
        W[WS_REACT + t] = r;
    }
    __syncthreads();

    // gconst = bp·w1 + reaction·w2 + ba ; Pconst = sum(bp)
    float c1 = 0.f, c2 = 0.f;
    if (t < D_PROJ) { c1 = bp[t] * wa[t] + react[t] * wa[D_PROJ + t]; c2 = bp[t]; }
    red[t] = c1; __syncthreads();
    for (int st = 128; st > 0; st >>= 1) { if (t < st) red[t] += red[t + st]; __syncthreads(); }
    float gtot = red[0];
    __syncthreads();
    red[t] = c2; __syncthreads();
    for (int st = 128; st > 0; st >>= 1) { if (t < st) red[t] += red[t + st]; __syncthreads(); }
    if (t == 0) { W[WS_CONST + 0] = gtot + ba[0]; W[WS_CONST + 1] = red[0]; }
}

// ------------------------------------------------------------------
// k1: stream prot_node once: copy to out (non-temporal) + per-row g[i], P[i].
//     One wave32 per row; 6250 blocks x 8 waves = 50000 rows exactly.
// ------------------------------------------------------------------
__global__ __launch_bounds__(256) void k1_stream(
    const float* __restrict__ prot, float* __restrict__ W, float* __restrict__ out)
{
    __shared__ float v1s[D_PROT], vps[D_PROT];
    const int t = threadIdx.x;
    v1s[t] = W[WS_V1 + t];
    vps[t] = W[WS_VP + t];
    __syncthreads();
    const float gconst = W[WS_CONST + 0];
    const float pconst = W[WS_CONST + 1];

    const int wid = t >> 5, lane = t & 31;
    const int row = blockIdx.x * 8 + wid;

    const v4f* src = (const v4f*)(prot + (size_t)row * D_PROT);
    v4f*       dst = (v4f*)(out + (size_t)row * D_PROT);
    const v4f x0 = src[lane * 2];
    const v4f x1 = src[lane * 2 + 1];
    __builtin_nontemporal_store(x0, dst + lane * 2);
    __builtin_nontemporal_store(x1, dst + lane * 2 + 1);

    const int bse = lane * 8;
    float sg = x0[0]*v1s[bse]   + x0[1]*v1s[bse+1] + x0[2]*v1s[bse+2] + x0[3]*v1s[bse+3]
             + x1[0]*v1s[bse+4] + x1[1]*v1s[bse+5] + x1[2]*v1s[bse+6] + x1[3]*v1s[bse+7];
    float sp = x0[0]*vps[bse]   + x0[1]*vps[bse+1] + x0[2]*vps[bse+2] + x0[3]*vps[bse+3]
             + x1[0]*vps[bse+4] + x1[1]*vps[bse+5] + x1[2]*vps[bse+6] + x1[3]*vps[bse+7];

    for (int off = 16; off > 0; off >>= 1) {
        sg += __shfl_down(sg, off, 32);
        sp += __shfl_down(sp, off, 32);
    }
    if (lane == 0) {
        W[WS_G + row] = sg + gconst;
        W[WS_P + row] = sp + pconst;
    }
}

// ------------------------------------------------------------------
// k1b: copy sub_node to out (non-temporal); emit int_index as floats
// ------------------------------------------------------------------
__global__ __launch_bounds__(256) void k1b_copy_sub(
    const float* __restrict__ sub, const int* __restrict__ idx, float* __restrict__ out)
{
    const int t = blockIdx.x * 256 + threadIdx.x;       // 500*256 = 128000 v4f's
    const v4f v = ((const v4f*)sub)[t];
    __builtin_nontemporal_store(v, (v4f*)(out + OUT_SUB) + t);
    if (blockIdx.x == 0 && threadIdx.x < N_INT)
        out[OUT_IDX + threadIdx.x] = (float)idx[threadIdx.x];
}

// ------------------------------------------------------------------
// k2a: per-window total = sum_k softmax(g[w..w+15])[k] * P[w+k];
//      block-level argmax (first-index tie-break)
// ------------------------------------------------------------------
__global__ __launch_bounds__(256) void k2a_windows(float* __restrict__ W)
{
    __shared__ float gs[256 + KERNEL_K], ps[256 + KERNEL_K];
    __shared__ float bval[256];
    __shared__ int   bidx[256];
    const int t = threadIdx.x;
    const int base = blockIdx.x * 256;

    int i = base + t;
    gs[t] = (i < N_PROT) ? W[WS_G + i] : 0.f;
    ps[t] = (i < N_PROT) ? W[WS_P + i] : 0.f;
    if (t < KERNEL_K) {
        int j = base + 256 + t;
        gs[256 + t] = (j < N_PROT) ? W[WS_G + j] : 0.f;
        ps[256 + t] = (j < N_PROT) ? W[WS_P + j] : 0.f;
    }
    __syncthreads();

    const int w = base + t;
    float val = -__builtin_inff();
    if (w < NUM_WIN) {
        float m = gs[t];
        for (int k = 1; k < KERNEL_K; ++k) m = fmaxf(m, gs[t + k]);
        float se = 0.f, sw = 0.f;
        for (int k = 0; k < KERNEL_K; ++k) {
            float e = __expf(gs[t + k] - m);
            se += e;
            sw += e * ps[t + k];
        }
        val = sw / se;
    }
    bval[t] = val;
    bidx[t] = (w < NUM_WIN) ? w : 0x7fffffff;
    __syncthreads();
    for (int st = 128; st > 0; st >>= 1) {
        if (t < st) {
            float v2 = bval[t + st]; int i2 = bidx[t + st];
            if (v2 > bval[t] || (v2 == bval[t] && i2 < bidx[t])) { bval[t] = v2; bidx[t] = i2; }
        }
        __syncthreads();
    }
    if (t == 0) {
        W[WS_BVAL + blockIdx.x] = bval[0];
        ((int*)(W + WS_INT))[blockIdx.x] = bidx[0];
    }
}

__global__ __launch_bounds__(256) void k2b_argmax(float* __restrict__ W, int nb)
{
    __shared__ float bval[256];
    __shared__ int   bidx[256];
    const int t = threadIdx.x;
    int* BI = (int*)(W + WS_INT);
    bval[t] = (t < nb) ? W[WS_BVAL + t] : -__builtin_inff();
    bidx[t] = (t < nb) ? BI[t] : 0x7fffffff;
    __syncthreads();
    for (int st = 128; st > 0; st >>= 1) {
        if (t < st) {
            float v2 = bval[t + st]; int i2 = bidx[t + st];
            if (v2 > bval[t] || (v2 == bval[t] && i2 < bidx[t])) { bval[t] = v2; bidx[t] = i2; }
        }
        __syncthreads();
    }
    if (t == 0) BI[256] = bidx[0];
}

// ------------------------------------------------------------------
// k3: WMMA projection of the winning 16x256 window through Wp (256x128),
//     softmax-weighted reduce -> top_score, then sub/prot deltas.
//     8 waves, wave w owns N-tile columns [16w, 16w+16).
//     K-loop fully unrolled: 8 v_wmma_f32_16x16x32_f16 with pipelined
//     load+cvt chains filling the WMMA D->A/B hazard slots.
// ------------------------------------------------------------------
__global__ __launch_bounds__(256) void k3_window_wmma(
    const float* __restrict__ prot, const float* __restrict__ Wp,
    const float* __restrict__ bp,
    const float* __restrict__ sub_out, const float* __restrict__ prot_out,
    float* __restrict__ W)
{
    __shared__ float gsh[KERNEL_K];
    __shared__ float ts[D_PROJ];
    const int t = threadIdx.x, wid = t >> 5, lane = t & 31;
    const int top = ((const int*)(W + WS_INT))[256];

    if (t < KERNEL_K) gsh[t] = W[WS_G + top + t];
    __syncthreads();

    // softmax over the 16 window gates; keep only this half-lane's 8 weights
    float m = gsh[0];
    for (int k = 1; k < KERNEL_K; ++k) m = fmaxf(m, gsh[k]);
    float se = 0.f;
    for (int k = 0; k < KERNEL_K; ++k) se += __expf(gsh[k] - m);
    const float inv = 1.f / se;
    const int hi = lane >> 4, mrow = lane & 15;
    float my_attn[8];
    for (int k = 0; k < 8; ++k) my_attn[k] = __expf(gsh[8 * hi + k] - m) * inv;

    // A: rows top..top+15 of prot_node (16x256), B: Wp columns [n0,n0+16)
    const int n0 = wid * 16;
    v8f acc = {};
#pragma unroll
    for (int kb = 0; kb < D_PROT / 32; ++kb) {
        v16h a, b;
        // A 16x32 f16 layout: lane(M=mrow, half=hi); e<8 -> K = kb*32+8*hi+e,
        //                                            e>=8 -> K = kb*32+8*hi+e+8
        const float* arow = prot + (size_t)(top + mrow) * D_PROT + kb * 32 + 8 * hi;
        const v4f a0 = ((const v4f*)arow)[0];
        const v4f a1 = ((const v4f*)arow)[1];
        const v4f a2 = ((const v4f*)(arow + 16))[0];
        const v4f a3 = ((const v4f*)(arow + 16))[1];
        a[0]=(_Float16)a0[0]; a[1]=(_Float16)a0[1]; a[2]=(_Float16)a0[2]; a[3]=(_Float16)a0[3];
        a[4]=(_Float16)a1[0]; a[5]=(_Float16)a1[1]; a[6]=(_Float16)a1[2]; a[7]=(_Float16)a1[3];
        a[8]=(_Float16)a2[0]; a[9]=(_Float16)a2[1]; a[10]=(_Float16)a2[2]; a[11]=(_Float16)a2[3];
        a[12]=(_Float16)a3[0]; a[13]=(_Float16)a3[1]; a[14]=(_Float16)a3[2]; a[15]=(_Float16)a3[3];
        // B 32x16 f16 layout: lane(N=mrow, half=hi); e<8 -> K = kb*32+8*hi+e,
        //                                            e>=8 -> K = kb*32+8*hi+e+8
        const float* brow = Wp + (size_t)(kb * 32 + 8 * hi) * D_PROJ + n0 + mrow;
#pragma unroll
        for (int e = 0; e < 8; ++e) {
            b[e]     = (_Float16)brow[(size_t)e * D_PROJ];
            b[e + 8] = (_Float16)brow[(size_t)(e + 16) * D_PROJ];
        }
        acc = __builtin_amdgcn_wmma_f32_16x16x32_f16(
                  false, a, false, b, (short)0, acc, false, false);
    }

    // acc[v] = proj[M = v + 8*hi][n0 + mrow] (pre-bias). Weighted reduce:
    const int n = n0 + mrow;
    const float bpn = bp[n];
    float part = 0.f;
#pragma unroll
    for (int v = 0; v < 8; ++v) part += my_attn[v] * (acc[v] + bpn);
    part += __shfl_xor(part, 16, 32);           // combine M=0..7 with M=8..15
    if (hi == 0) ts[n] = part + W[WS_REACT + n];
    __syncthreads();

    // deltas: top_score (128) @ sub_out(128x256), @ prot_out(128x256)
    float sd = 0.f, pd = 0.f;
    for (int d = 0; d < D_PROJ; ++d) {
        const float x = ts[d];
        sd += x * sub_out[(size_t)d * D_SUB + t];
        pd += x * prot_out[(size_t)d * D_PROT + t];
    }
    W[WS_SDELTA + t] = sd;
    W[WS_PDELTA + t] = pd;
}

// ------------------------------------------------------------------
// k4: fix-ups. Blocks 0..15: prot rows top+b. Blocks 16..79: sub rows idx[b-16].
//     Writes derive from ORIGINAL inputs -> idempotent for duplicate indices.
// ------------------------------------------------------------------
__global__ __launch_bounds__(256) void k4_fixup(
    const float* __restrict__ prot, const float* __restrict__ sub,
    const int* __restrict__ idx, const float* __restrict__ W, float* __restrict__ out)
{
    const int t = threadIdx.x, b = blockIdx.x;
    const int top = ((const int*)(W + WS_INT))[256];
    if (b < KERNEL_K) {
        const int r = top + b;
        out[(size_t)r * D_PROT + t] = prot[(size_t)r * D_PROT + t] + W[WS_PDELTA + t];
    } else {
        const int j = idx[b - KERNEL_K];
        out[OUT_SUB + (size_t)j * D_SUB + t] = sub[(size_t)j * D_SUB + t] + W[WS_SDELTA + t];
    }
}

// ------------------------------------------------------------------
extern "C" void kernel_launch(void* const* d_in, const int* in_sizes, int n_in,
                              void* d_out, int out_size, void* d_ws, size_t ws_size,
                              hipStream_t stream)
{
    const float* prot     = (const float*)d_in[0];
    const float* sub      = (const float*)d_in[1];
    const int*   idx      = (const int*)  d_in[2];
    const float* Wp       = (const float*)d_in[3];
    const float* bp       = (const float*)d_in[4];
    const float* Ws       = (const float*)d_in[5];
    const float* bs       = (const float*)d_in[6];
    const float* wa       = (const float*)d_in[7];
    const float* ba       = (const float*)d_in[8];
    const float* sub_out  = (const float*)d_in[9];
    const float* prot_out = (const float*)d_in[10];
    float* out = (float*)d_out;
    float* W   = (float*)d_ws;

    const int nb2 = (NUM_WIN + 255) / 256;   // 196

    k0_precompute  <<<1,                 256, 0, stream>>>(sub, idx, Wp, bp, Ws, bs, wa, ba, W);
    k1_stream      <<<N_PROT / 8,        256, 0, stream>>>(prot, W, out);
    k1b_copy_sub   <<<(N_SUB*D_SUB/4)/256, 256, 0, stream>>>(sub, idx, out);
    k2a_windows    <<<nb2,               256, 0, stream>>>(W);
    k2b_argmax     <<<1,                 256, 0, stream>>>(W, nb2);
    k3_window_wmma <<<1,                 256, 0, stream>>>(prot, Wp, bp, sub_out, prot_out, W);
    k4_fixup       <<<KERNEL_K + N_INT,  256, 0, stream>>>(prot, sub, idx, W, out);
}
